// FastNeuralMemory_89687507076228
// MI455X (gfx1250) — matrix operand order repaired and checked by reading, not compiled, not run on hardware
//
#include <hip/hip_runtime.h>
#include <hip/hip_bf16.h>

// ---------------------------------------------------------------------------
// FastNeuralMemory for MI455X (gfx1250, wave32, WMMA)
//
//  1) f32 -> bf16 conversion of x and all weight matrices
//  2) WMMA bf16 GEMMs: K = x@Wk, V = x@Wv, Q = x@Wq, gpre = x@Wg + bg
//     - A tile staged via global_load_async_to_lds_b128 (ASYNCcnt path)
//     - B tile staged TRANSPOSED in LDS so fragments load as 2x ds_load_b128
//  3) Per-head LayerNorm (warp-per-row); k additionally L2-normalized -> kn
//  4) Sequential delta-rule scan: 1 workgroup per (b,h) chain, 64x64 memory
//     + momentum in LDS (row stride 65, conflict-free), async vector loads
//  5) outs -> bf16, WMMA GEMM outs@Wo, out = sigmoid(x@Wg+bg) * (outs@Wo)
// ---------------------------------------------------------------------------

typedef __bf16 bf16;
typedef float    v8f    __attribute__((ext_vector_type(8)));
typedef bf16     v16bf  __attribute__((ext_vector_type(16)));
typedef bf16     bf16x4 __attribute__((ext_vector_type(4)));
typedef unsigned uint4v __attribute__((ext_vector_type(4)));   // 16B, trivial

#define NHEADS 8
#define HDIM   64

// Async 16B global -> LDS copy (per-lane), tracked by ASYNCcnt.
__device__ __forceinline__ void async_g2l_b128(void* lds, const void* gsrc) {
    asm volatile("global_load_async_to_lds_b128 %0, %1, off"
                 :: "v"((unsigned)(unsigned long long)lds),
                    "v"((unsigned long long)gsrc)
                 : "memory");
}
__device__ __forceinline__ void wait_asynccnt0() {
    asm volatile("s_wait_asynccnt 0x0" ::: "memory");
}

// -------------------------------- convert ---------------------------------
__global__ __launch_bounds__(256) void cvt_f32_bf16(
    const float* __restrict__ in, bf16* __restrict__ out, long n)
{
    long i = ((long)blockIdx.x * blockDim.x + threadIdx.x) * 4;
    if (i + 3 < n) {
        float4 f = *(const float4*)(in + i);
        bf16x4 o;
        o.x = (bf16)f.x; o.y = (bf16)f.y; o.z = (bf16)f.z; o.w = (bf16)f.w;
        *(bf16x4*)(out + i) = o;
    } else {
        for (; i < n; ++i) out[i] = (bf16)in[i];
    }
}

// ----------------------------- WMMA bf16 GEMM ------------------------------
// C(MxN) = A(MxK) @ B(KxN) [+ bias], fp32 accumulate.
// Block: 128 threads (4 waves). Block tile 64x64, K-step 32.
// Wave w computes rows [16w,16w+16) x 64 cols = 4 x (16x16) WMMA tiles.
// M,N,K multiples of 64/64/32 (true for all launches here).
#define BM 64
#define BN 64
#define BK 32
#define LSTR (BK + 8)   // 40 bf16 = 80 B rows: 16B-aligned, conflict-friendly

__global__ __launch_bounds__(128) void gemm_bf16_wmma(
    const bf16* __restrict__ A, const bf16* __restrict__ Bm,
    float* __restrict__ C, const float* __restrict__ bias,
    int M, int N, int K)
{
    __shared__ __align__(16) bf16 As[BM][LSTR];   // row-major A tile
    __shared__ __align__(16) bf16 Bt[BN][LSTR];   // TRANSPOSED B tile: [col][k]

    const int tid  = threadIdx.x;
    const int wave = tid >> 5;
    const int lane = tid & 31;
    const int half = lane >> 4;
    const int l16  = lane & 15;

    const int rowBase = blockIdx.y * BM;
    const int colBase = blockIdx.x * BN;

    v8f acc[4] = {};

    for (int k0 = 0; k0 < K; k0 += BK) {
        // ---- A tile: 64x32 bf16 = 256 x 16B chunks, async copy, 2/thread
        #pragma unroll
        for (int c = 0; c < 2; ++c) {
            int idx = tid + c * 128;
            int r = idx >> 2, cc = (idx & 3) << 3;
            async_g2l_b128(&As[r][cc],
                           A + (size_t)(rowBase + r) * K + k0 + cc);
        }
        // ---- B tile: register transpose into Bt[col][k]
        // 2 chunks/thread, each chunk = 8 columns x one K-row-pair
        #pragma unroll
        for (int c = 0; c < 2; ++c) {
            int idx = tid + c * 128;
            int p = idx >> 3, cc = (idx & 7) << 3;    // p = K pair 0..31
            union { uint4v q; unsigned short s[8]; } r0, r1;
            r0.q = *(const uint4v*)(Bm + (size_t)(k0 + 2*p)     * N + colBase + cc);
            r1.q = *(const uint4v*)(Bm + (size_t)(k0 + 2*p + 1) * N + colBase + cc);
            #pragma unroll
            for (int i = 0; i < 8; ++i) {
                unsigned pk = (unsigned)r0.s[i] | ((unsigned)r1.s[i] << 16);
                *(unsigned*)&Bt[cc + i][2*p] = pk;    // K pair contiguous
            }
        }
        wait_asynccnt0();
        __syncthreads();

        // ---- A fragment (16x32): lanes 0-15 K {0..7,16..23},
        //      lanes 16-31 K {8..15,24..31}; two contiguous 16B runs.
        union { v16bf v; uint4v q[2]; } af;
        const bf16* arow = &As[16 * wave + l16][half ? 8 : 0];
        af.q[0] = *(const uint4v*)(arow);
        af.q[1] = *(const uint4v*)(arow + 16);

        // ---- 4 column tiles; B fragment = same K order as A from Bt[col][*]
        #pragma unroll
        for (int c4 = 0; c4 < 4; ++c4) {
            union { v16bf v; uint4v q[2]; } bfr;
            const bf16* bcol = &Bt[16 * c4 + l16][half ? 8 : 0];
            bfr.q[0] = *(const uint4v*)(bcol);
            bfr.q[1] = *(const uint4v*)(bcol + 16);
            acc[c4] = __builtin_amdgcn_wmma_f32_16x16x32_bf16(
                false, af.v, false, bfr.v, (short)0, acc[c4], false, false);
        }
        __syncthreads();
    }

    // Epilogue: C/D layout — VGPR i holds M=i (lanes 0-15) / M=8+i (16-31)
    #pragma unroll
    for (int c = 0; c < 4; ++c) {
        int gc = colBase + 16 * c + l16;
        float bvv = bias ? bias[gc] : 0.0f;
        #pragma unroll
        for (int i = 0; i < 8; ++i) {
            int gr = rowBase + 16 * wave + (half ? 8 : 0) + i;
            C[(size_t)gr * N + gc] = acc[c][i] + bvv;
        }
    }
}

// --------------------------- LayerNorm (HD=64) -----------------------------
__global__ __launch_bounds__(128) void ln64(
    float* __restrict__ data, const float* __restrict__ g,
    const float* __restrict__ b, long rows, int l2norm)
{
    int warp = threadIdx.x >> 5, lane = threadIdx.x & 31;
    long row = (long)blockIdx.x * 4 + warp;
    if (row >= rows) return;
    float* p = data + row * 64;
    float x0 = p[lane], x1 = p[lane + 32];

    float s = x0 + x1;
    for (int o = 16; o > 0; o >>= 1) s += __shfl_xor(s, o, 32);
    float mu = s * (1.0f / 64.0f);
    float d0 = x0 - mu, d1 = x1 - mu;
    float vs = d0 * d0 + d1 * d1;
    for (int o = 16; o > 0; o >>= 1) vs += __shfl_xor(vs, o, 32);
    float inv = rsqrtf(vs * (1.0f / 64.0f) + 1e-5f);
    float y0 = d0 * inv * g[lane] + b[lane];
    float y1 = d1 * inv * g[lane + 32] + b[lane + 32];
    if (l2norm) {
        float n2 = y0 * y0 + y1 * y1;
        for (int o = 16; o > 0; o >>= 1) n2 += __shfl_xor(n2, o, 32);
        float sc = 1.0f / (sqrtf(n2) + 1e-6f);
        y0 *= sc; y1 *= sc;
    }
    p[lane] = y0; p[lane + 32] = y1;
}

// ------------------------------- scan kernel -------------------------------
// One workgroup (256 threads) per (b,h) chain; 64x64 mem/momentum in LDS.
#define MSTR 65

__global__ __launch_bounds__(256) void scan_fnm(
    const float* __restrict__ KN, const float* __restrict__ V,
    const float* __restrict__ Q,
    const float* __restrict__ mem_in, const float* __restrict__ mbuf_in,
    const float* __restrict__ lr_scale, const float* __restrict__ mom_scale,
    float* __restrict__ OUTS, float* __restrict__ mem_out,
    float* __restrict__ mbuf_out, int S)
{
    __shared__ float mem_s[64 * MSTR];
    __shared__ float mbuf_s[64 * MSTR];
    __shared__ __align__(16) float kn_s[64];
    __shared__ __align__(16) float v_s[64];
    __shared__ __align__(16) float q_s[64];
    __shared__ float pr_s[64];

    const int tid = threadIdx.x;
    const int bh  = blockIdx.x;          // b*NHEADS + h
    const int h   = bh & (NHEADS - 1);
    const int b   = bh >> 3;

    const float lr  = 0.2f / (1.0f + __expf(-lr_scale[h]));   // sigmoid*0.1*2
    const float mom = 1.8f / (1.0f + __expf(-mom_scale[h]));  // sigmoid*0.9*2

    const size_t st = (size_t)bh * (HDIM * HDIM);
    for (int e = tid; e < HDIM * HDIM; e += 256) {
        int r = e >> 6, c = e & 63;
        mem_s[r * MSTR + c]  = mem_in[st + e];
        mbuf_s[r * MSTR + c] = mbuf_in[st + e];
    }
    __syncthreads();

    const int d = tid >> 2;   // 0..63 row index
    const int j = tid & 3;    // 0..3  16-wide K chunk
    float* mr  = &mem_s[d * MSTR + j * 16];
    float* mbr = &mbuf_s[d * MSTR + j * 16];

    size_t base = ((size_t)b * S) * (NHEADS * HDIM) + (size_t)h * HDIM;

    for (int t = 0; t < S; ++t) {
        // async-stage kn/v/q (each 64 floats = 16 x 16B) into LDS
        if (tid < 48) {
            int which = tid >> 4;            // 0=kn 1=v 2=q
            int l = (tid & 15) << 2;         // float offset, 16B chunks
            const float* g = (which == 0 ? KN : which == 1 ? V : Q) + base + l;
            float* lds = (which == 0 ? kn_s : which == 1 ? v_s : q_s) + l;
            async_g2l_b128(lds, g);
        }
        if (t + 1 < S && tid < 64) {
            __builtin_prefetch(KN + base + 512 + tid, 0, 0);
            __builtin_prefetch(Q  + base + 512 + tid, 0, 0);
        }
        wait_asynccnt0();
        __syncthreads();

        float so = 0.0f, sp = 0.0f;
        #pragma unroll
        for (int c = 0; c < 16; ++c) {
            float m = mr[c];
            so += m * q_s[j * 16 + c];
            sp += m * kn_s[j * 16 + c];
        }
        so += __shfl_xor(so, 1, 32); so += __shfl_xor(so, 2, 32);
        sp += __shfl_xor(sp, 1, 32); sp += __shfl_xor(sp, 2, 32);
        if (j == 0) {
            pr_s[d] = sp;
            OUTS[base + d] = so;
        }
        __syncthreads();

        const float pd = pr_s[d];
        const float sd = pd - v_s[d];
        #pragma unroll
        for (int c = 0; c < 16; ++c) {
            float kn = kn_s[j * 16 + c];
            float mb = mom * mbr[c] + sd * kn;
            mbr[c] = mb;
            mr[c]  = mr[c] - pd * kn - lr * mb;
        }
        __syncthreads();
        base += NHEADS * HDIM;
    }

    for (int e = tid; e < HDIM * HDIM; e += 256) {
        int r = e >> 6, c = e & 63;
        mem_out[st + e]  = mem_s[r * MSTR + c];
        mbuf_out[st + e] = mbuf_s[r * MSTR + c];
    }
}

// ----------------------------- gate epilogue -------------------------------
__global__ __launch_bounds__(256) void gate_mul(
    const float* __restrict__ gpre, const float* __restrict__ oraw,
    float* __restrict__ out, long n)
{
    long i = (long)blockIdx.x * blockDim.x + threadIdx.x;
    if (i < n) {
        float g = 1.0f / (1.0f + __expf(-gpre[i]));
        out[i] = g * oraw[i];
    }
}

// ------------------------------- launcher ----------------------------------
extern "C" void kernel_launch(void* const* d_in, const int* in_sizes, int n_in,
                              void* d_out, int out_size, void* d_ws, size_t ws_size,
                              hipStream_t stream)
{
    (void)in_sizes; (void)n_in; (void)out_size; (void)ws_size;

    const float* x    = (const float*)d_in[0];
    const float* mem0 = (const float*)d_in[1];
    const float* mb0  = (const float*)d_in[2];
    const float* Wk   = (const float*)d_in[3];
    const float* Wv   = (const float*)d_in[4];
    const float* Wq   = (const float*)d_in[5];
    const float* Wo   = (const float*)d_in[6];
    const float* gk   = (const float*)d_in[7];
    const float* bk   = (const float*)d_in[8];
    const float* gv   = (const float*)d_in[9];
    const float* bvp  = (const float*)d_in[10];
    const float* gq   = (const float*)d_in[11];
    const float* bq   = (const float*)d_in[12];
    const float* lrs  = (const float*)d_in[13];
    const float* moms = (const float*)d_in[14];
    const float* Wg   = (const float*)d_in[15];
    const float* bg   = (const float*)d_in[16];

    float* out = (float*)d_out;

    const int  Bsz = 4, S = 2048, DIM = 1024, NHHD = NHEADS * HDIM; // 512
    const long Mrows = (long)Bsz * S;                               // 8192

    char* ws = (char*)d_ws;
    size_t off = 0;
    auto alloc = [&](size_t bytes) -> void* {
        void* p = ws + off;
        off += (bytes + 255) & ~(size_t)255;
        return p;
    };

    bf16*  xb    = (bf16*) alloc((size_t)Mrows * DIM  * sizeof(bf16));
    bf16*  wkb   = (bf16*) alloc((size_t)DIM * NHHD   * sizeof(bf16));
    bf16*  wvb   = (bf16*) alloc((size_t)DIM * NHHD   * sizeof(bf16));
    bf16*  wqb   = (bf16*) alloc((size_t)DIM * NHHD   * sizeof(bf16));
    bf16*  wob   = (bf16*) alloc((size_t)NHHD * DIM   * sizeof(bf16));
    bf16*  wgb   = (bf16*) alloc((size_t)DIM * DIM    * sizeof(bf16));
    float* Kp    = (float*)alloc((size_t)Mrows * NHHD * sizeof(float));
    float* Vp    = (float*)alloc((size_t)Mrows * NHHD * sizeof(float));
    float* Qp    = (float*)alloc((size_t)Mrows * NHHD * sizeof(float));
    float* outs  = (float*)alloc((size_t)Mrows * NHHD * sizeof(float));
    bf16*  outsb = (bf16*) alloc((size_t)Mrows * NHHD * sizeof(bf16));
    float* oraw  = (float*)alloc((size_t)Mrows * DIM  * sizeof(float));
    float* gpre  = (float*)alloc((size_t)Mrows * DIM  * sizeof(float));

    auto cvt = [&](const float* src, bf16* dst, long n) {
        unsigned nb = (unsigned)((n / 4 + 255) / 256);
        cvt_f32_bf16<<<dim3(nb), dim3(256), 0, stream>>>(src, dst, n);
    };
    cvt(x,  xb,  Mrows * DIM);
    cvt(Wk, wkb, (long)DIM * NHHD);
    cvt(Wv, wvb, (long)DIM * NHHD);
    cvt(Wq, wqb, (long)DIM * NHHD);
    cvt(Wo, wob, (long)NHHD * DIM);
    cvt(Wg, wgb, (long)DIM * DIM);

    auto gemm = [&](const bf16* A, const bf16* Bm, float* C, const float* bias,
                    int M, int N, int K) {
        gemm_bf16_wmma<<<dim3(N / 64, M / 64), dim3(128), 0, stream>>>(
            A, Bm, C, bias, M, N, K);
    };
    gemm(xb, wkb, Kp,   nullptr, (int)Mrows, NHHD, DIM);
    gemm(xb, wvb, Vp,   nullptr, (int)Mrows, NHHD, DIM);
    gemm(xb, wqb, Qp,   nullptr, (int)Mrows, NHHD, DIM);
    gemm(xb, wgb, gpre, bg,      (int)Mrows, DIM,  DIM);

    long rows = Mrows * NHEADS;   // 65536 LayerNorm rows of width 64
    unsigned lnb = (unsigned)((rows + 3) / 4);
    ln64<<<dim3(lnb), dim3(128), 0, stream>>>(Kp, gk, bk,  rows, 1); // -> kn
    ln64<<<dim3(lnb), dim3(128), 0, stream>>>(Vp, gv, bvp, rows, 0);
    ln64<<<dim3(lnb), dim3(128), 0, stream>>>(Qp, gq, bq,  rows, 0);

    float* memOut  = out + (size_t)Mrows * DIM;               // 8,388,608
    float* mbufOut = memOut + (size_t)Bsz * NHEADS * HDIM * HDIM;
    scan_fnm<<<dim3(Bsz * NHEADS), dim3(256), 0, stream>>>(
        Kp, Vp, Qp, mem0, mb0, lrs, moms, outs, memOut, mbufOut, S);

    cvt(outs, outsb, Mrows * NHHD);
    gemm(outsb, wob, oraw, nullptr, (int)Mrows, DIM, NHHD);

    long n = Mrows * DIM;
    gate_mul<<<dim3((unsigned)((n + 255) / 256)), dim3(256), 0, stream>>>(
        gpre, oraw, out, n);
}
